// DCTNMnistModel_35613868818920
// MI455X (gfx1250) — compile-verified, matrix operand order checked by reading, not run on hardware
//
#include <hip/hip_runtime.h>
#include <hip/hip_bf16.h>

// ---------------------------------------------------------------------------
// DCTN MNIST tensor-network forward for MI455X (gfx1250).
//
// ~0.15 TFLOP total, all operands (<4MB) L2-resident (192MB L2, 23.3TB/s HBM)
// -> purely compute bound. Strategy:
//   * trace reorder:  out_o = <Mid_o, (R*L)^T>,  L = M0..M3, R = M5..M8
//   * coefficient expansion: acc*M_t = [coef_c(pix)*acc] x [stacked cores]
//     -> shared-B GEMMs across pixels -> v_wmma_f32_16x16x32_bf16
//   * side cores staged into LDS via the Tensor Data Mover, s_wait_tensorcnt
//   * per-pixel P=R*L and the mid contraction in f32 VALU
// Templated on NC so WMMA K-loops fully unroll; B fragments preloaded per
// column tile (3x reuse); flat row*24+col addressing (no divisions) in the
// hot A-build / writeback paths.
// ---------------------------------------------------------------------------

#define DD     24
#define NP     8            // pixels per workgroup
#define KPAD   96           // expanded K (4 coef slots * 24), zero padded
#define ROWS   (NP*DD)      // 192 GEMM rows per workgroup
#define BLOCK  128          // 4 waves

typedef __attribute__((ext_vector_type(16))) __bf16       v16bf;
typedef __attribute__((ext_vector_type(8)))  float        v8f;
typedef __attribute__((ext_vector_type(4)))  unsigned int v4u;
typedef __attribute__((ext_vector_type(8)))  int          v8i_t;
typedef __attribute__((ext_vector_type(4)))  int          v4i_t;

// LDS layout (dynamic, 184448 B < 320KB/WGP)
#define OFF_STAGE 0
#define SZ_STAGE  (8*4*576*4)
#define OFF_BPAN  (OFF_STAGE + SZ_STAGE) // 73728
#define SZ_BPAN   (6*32*KPAD*2)
#define OFF_AEXP  (OFF_BPAN + SZ_BPAN)   // 110592
#define SZ_AEXP   (ROWS*KPAD*2)
#define OFF_ACC   (OFF_AEXP + SZ_AEXP)   // 147456
#define SZ_ACC    (NP*576*4)
#define OFF_LSV   (OFF_ACC + SZ_ACC)     // 165888
#define OFF_COEF  (OFF_LSV + SZ_ACC)     // 184320
#define LDS_TOTAL (OFF_COEF + NP*4*4)    // 184448

// POS_A / POS_B window offsets
__device__ __constant__ int POS_DH[2][9] = {{0,0,0,1,1,1,2,2,2},{0,1,2,2,1,0,0,1,2}};
__device__ __constant__ int POS_DW[2][9] = {{0,1,2,2,1,0,0,1,2},{0,0,0,1,1,1,2,2,2}};

template<int NC>
__device__ __forceinline__ void tn_coefs(int tid, long long gpix0, int hw,
                                         int Hin, int Win, int Wout,
                                         const float* ch0, const float* ch1,
                                         int dh, int dw, float* COEF)
{
    if (tid < NP) {
        long long gp = gpix0 + tid;
        int b  = (int)(gp / hw);
        int rr = (int)(gp - (long long)b * hw);
        int h  = rr / Wout, w = rr - (rr / Wout) * Wout;
        long long base = (((long long)b * Hin + (h + dh)) * (long long)Win + (w + dw)) * 2;
        float a0 = ch0[base], a1 = ch0[base + 1];
        if (NC == 4) {
            float b0 = ch1[base], b1 = ch1[base + 1];
            COEF[tid*4+0] = a0*b0; COEF[tid*4+1] = a0*b1;
            COEF[tid*4+2] = a1*b0; COEF[tid*4+3] = a1*b1;
        } else {
            COEF[tid*4+0] = a0; COEF[tid*4+1] = a1;
            COEF[tid*4+2] = 0.f; COEF[tid*4+3] = 0.f;
        }
    }
}

template<int NC>
__device__ __forceinline__ void tn_init_acc(int tid, int coreIdx,
                                            const float* STAGE, const float* COEF, float* ACC)
{
    for (int i = tid; i < NP*576; i += BLOCK) {
        int p = i / 576, e = i - p*576;
        const float* cs = STAGE + (coreIdx * NC) * 576 + e;
        float s = 0.f;
        #pragma unroll
        for (int c = 0; c < NC; ++c) s += COEF[p*4+c] * cs[c*576];
        ACC[i] = s;
    }
}

// One chain step: acc <- acc * M_t via coefficient-expanded shared-B WMMA GEMM.
template<int NC>
__device__ __forceinline__ void tn_chain_step(int tid, int lane, int sPanel,
                                              float* ACC, __bf16* AEXP,
                                              const __bf16* BPAN, const float* COEF)
{
    constexpr int KCH = (NC == 2) ? 2 : 3;   // K chunks of 32 over K = NC*24

    // Build A[row, (c,k)] = coef_c(pix) * acc[row,k]; note acc[row,k] == ACC[i]
    for (int i = tid; i < ROWS*DD; i += BLOCK) {
        int row = i / DD, k = i - row*DD;
        int p   = i / 576;
        float av = ACC[i];
        __bf16* arow = AEXP + row*KPAD + k;
        #pragma unroll
        for (int c = 0; c < NC; ++c) arow[c*DD] = (__bf16)(av * COEF[p*4+c]);
    }
    __syncthreads();

    const int wave  = tid >> 5;
    const int hhalf = (lane & 16) ? 8 : 0;       // ISA 16-bit fragment K offset
    #pragma unroll
    for (int ct = 0; ct < 2; ++ct) {             // 2 column tiles (r = 0..23 pad 32)
        // Preload B fragments once per column tile; reused by 3 row tiles.
        v16bf bfrag[KCH];
        const int bcol = ct*16 + (lane & 15);
        #pragma unroll
        for (int kc = 0; kc < KCH; ++kc) {
            const __bf16* bp = BPAN + (sPanel*32 + bcol)*KPAD + kc*32 + hhalf;
            union { uint4 u[2]; v16bf v; } b_;
            b_.u[0] = *(const uint4*)(bp);
            b_.u[1] = *(const uint4*)(bp + 16);
            bfrag[kc] = b_.v;
        }
        #pragma unroll
        for (int rr = 0; rr < 3; ++rr) {         // 12 row tiles / 4 waves
            const int rt = wave + rr*4;
            const int arow = rt*16 + (lane & 15);
            v8f cfrag = {0.f,0.f,0.f,0.f,0.f,0.f,0.f,0.f};
            #pragma unroll
            for (int kc = 0; kc < KCH; ++kc) {
                const __bf16* ap = AEXP + arow*KPAD + kc*32 + hhalf;
                union { uint4 u[2]; v16bf v; } af;
                af.u[0] = *(const uint4*)(ap);
                af.u[1] = *(const uint4*)(ap + 16);
                cfrag = __builtin_amdgcn_wmma_f32_16x16x32_bf16(
                    false, af.v, false, bfrag[kc], (short)0, cfrag, false, false);
            }
            // C/D f32 layout: VGPR v -> row rt*16 + hbase + v, col = ct*16+lane15.
            // Dest index (pix*576 + l*24 + col) == row*24 + col  -> base + v*24.
            const int colb = ct*16 + (lane & 15);
            if (colb < DD) {
                float* dst = ACC + (rt*16 + ((lane & 16) ? 8 : 0))*DD + colb;
                #pragma unroll
                for (int v = 0; v < 8; ++v) dst[v*DD] = cfrag[v];
            }
        }
    }
    __syncthreads();
}

template<int NC, int MIDO, int FINAL>
__global__ __launch_bounds__(BLOCK)
void tn_string_kernel(const float* __restrict__ ch0, const float* __restrict__ ch1,
                      const float* __restrict__ side, const float* __restrict__ mid,
                      float* __restrict__ out,
                      int Hin, int Win, int posSel, float outScale)
{
    extern __shared__ char lds[];
    float*  STAGE = (float*)(lds + OFF_STAGE);
    __bf16* BPAN  = (__bf16*)(lds + OFF_BPAN);
    __bf16* AEXP  = (__bf16*)(lds + OFF_AEXP);
    float*  PBUF  = (float*)(lds + OFF_AEXP);   // alias: reused after chain done
    float*  ACC   = (float*)(lds + OFF_ACC);
    float*  LSV   = (float*)(lds + OFF_LSV);
    float*  COEF  = (float*)(lds + OFF_COEF);

    const int tid  = threadIdx.x;
    const int lane = tid & 31;
    const int wave = tid >> 5;
    const int Hout = Hin - 2, Wout = Win - 2;
    const int hw   = Hout * Wout;
    const long long gpix0 = (long long)blockIdx.x * NP;
    constexpr int coreStride = NC * 576;

    __builtin_prefetch(mid, 0, 1);   // global_prefetch_b8: mid cores are L2-hot

    // --- Stage all 8 side cores (f32) into LDS via the Tensor Data Mover ----
#if __has_builtin(__builtin_amdgcn_tensor_load_to_lds)
    if (wave == 0) {
        unsigned long long ga = (unsigned long long)(size_t)side;
        v4u g0;
        g0.x = 1u;                                      // count = 1 (valid D#)
        g0.y = (unsigned)OFF_STAGE;                     // lds_addr
        g0.z = (unsigned)(ga & 0xFFFFFFFFull);          // global_addr[31:0]
        g0.w = (unsigned)((ga >> 32) & 0x01FFFFFFull)   // global_addr[56:32]
             | (2u << 30);                              // type = 2 ("image")
        constexpr int t0 = coreStride;                  // elems per core (4B each)
        v8i_t g1;
        g1[0] = (2 << 16);                              // data_size = 4 bytes
        g1[1] = (t0 & 0xFFFF) << 16;                    // tensor_dim0[15:0]
        g1[2] = ((t0 >> 16) & 0xFFFF) | (8 << 16);      // dim0[31:16] | tensor_dim1=8
        g1[3] = (t0 & 0xFFFF) << 16;                    // tile_dim0 = t0
        g1[4] = 8;                                      // tile_dim1 = 8 cores
        g1[5] = t0;                                     // tensor_dim0_stride
        g1[6] = 0;
        g1[7] = 0;
        v4i_t gz4 = {0,0,0,0};                          // groups 2/3 unused (2D)
        v8i_t gz8 = {0,0,0,0,0,0,0,0};
        __builtin_amdgcn_tensor_load_to_lds(g0, g1, gz4, gz4, gz8, 0);
        __builtin_amdgcn_s_wait_tensorcnt(0);
    }
#else
    for (int i = tid; i < 8*coreStride; i += BLOCK) STAGE[i] = side[i];
#endif
    __syncthreads();

    // --- Build col-major bf16 B panels for the 6 WMMA steps (cores 1,2,3,5,6,7)
    {
        const int SC[6] = {1,2,3,5,6,7};
        for (int idx = tid; idx < 6*32*KPAD; idx += BLOCK) {
            int s   = idx / (32*KPAD);
            int rem = idx - s*(32*KPAD);
            int col = rem / KPAD;
            int ck  = rem - col*KPAD;
            int c   = ck / DD;
            int k   = ck - c*DD;
            float v = 0.f;
            if (col < DD && c < NC)
                v = STAGE[(SC[s]*NC + c)*576 + k*DD + col];
            BPAN[idx] = (__bf16)v;
        }
        for (int i = tid; i < ROWS*KPAD; i += BLOCK) AEXP[i] = (__bf16)0.f; // K pad
    }
    __syncthreads();

    // --- L = M0 * M1 * M2 * M3 ---------------------------------------------
    tn_coefs<NC>(tid, gpix0, hw, Hin, Win, Wout, ch0, ch1,
                 POS_DH[posSel][0], POS_DW[posSel][0], COEF);
    __syncthreads();
    tn_init_acc<NC>(tid, 0, STAGE, COEF, ACC);
    __syncthreads();
    #pragma unroll
    for (int s = 0; s < 3; ++s) {
        tn_coefs<NC>(tid, gpix0, hw, Hin, Win, Wout, ch0, ch1,
                     POS_DH[posSel][s+1], POS_DW[posSel][s+1], COEF);
        __syncthreads();
        tn_chain_step<NC>(tid, lane, s, ACC, AEXP, BPAN, COEF);
    }
    for (int i = tid; i < NP*576; i += BLOCK) LSV[i] = ACC[i];
    __syncthreads();

    // --- R = M5 * M6 * M7 * M8 ---------------------------------------------
    tn_coefs<NC>(tid, gpix0, hw, Hin, Win, Wout, ch0, ch1,
                 POS_DH[posSel][5], POS_DW[posSel][5], COEF);
    __syncthreads();
    tn_init_acc<NC>(tid, 4, STAGE, COEF, ACC);
    __syncthreads();
    #pragma unroll
    for (int s = 3; s < 6; ++s) {
        tn_coefs<NC>(tid, gpix0, hw, Hin, Win, Wout, ch0, ch1,
                     POS_DH[posSel][s+3], POS_DW[posSel][s+3], COEF);
        __syncthreads();
        tn_chain_step<NC>(tid, lane, s, ACC, AEXP, BPAN, COEF);
    }

    // --- P = R * L per pixel (f32 VALU, D^3 = 13.8k MAC/pixel) -------------
    for (int i = tid; i < NP*576; i += BLOCK) {
        int e = i % 576;            // i = p*576 + r*24 + k
        int k = e % DD;
        const float* Rw = ACC + (i - k);        // p*576 + r*24
        const float* Lw = LSV + (i - e) + k;    // p*576 + k
        float s = 0.f;
        #pragma unroll 8
        for (int j = 0; j < DD; ++j) s += Rw[j] * Lw[j*DD];
        PBUF[i] = s;
    }
    __syncthreads();
    tn_coefs<NC>(tid, gpix0, hw, Hin, Win, Wout, ch0, ch1,
                 POS_DH[posSel][4], POS_DW[posSel][4], COEF);
    __syncthreads();

    // --- out_o = sum_c coef_c(pos4) * <MidCore[c](:,:,o), P^T> -------------
    for (int idx = tid; idx < NP*MIDO; idx += BLOCK) {
        int p = idx / MIDO, o = idx - p*MIDO;
        const float* Pp = PBUF + p*576;
        float acc_o = 0.f;
        #pragma unroll
        for (int c = 0; c < NC; ++c) {
            const float* mc = mid + (long long)(c*576)*MIDO + o;
            float t = 0.f;
            for (int k = 0; k < DD; ++k)
                #pragma unroll 8
                for (int r = 0; r < DD; ++r)
                    t += mc[(k*DD + r)*MIDO] * Pp[r*DD + k];
            acc_o += COEF[p*4+c] * t;
        }
        long long gp = gpix0 + p;
        int b  = (int)(gp / hw);
        int rr = (int)(gp - (long long)b * hw);
        if (FINAL) atomicAdd(out + b*MIDO + o, acc_o * outScale);
        else       out[((long long)b*hw + rr)*MIDO + o] = acc_o;
    }
}

__global__ void tn_embed_kernel(const float* __restrict__ x, float* __restrict__ q, int n)
{
    int i = blockIdx.x * blockDim.x + threadIdx.x;
    if (i < n) { float v = x[i]; q[2*i+0] = sinf(v); q[2*i+1] = cosf(v); }
}

__global__ void tn_zero_kernel(float* p, int n)
{
    int i = blockIdx.x * blockDim.x + threadIdx.x;
    if (i < n) p[i] = 0.f;
}

extern "C" void kernel_launch(void* const* d_in, const int* in_sizes, int n_in,
                              void* d_out, int out_size, void* d_ws, size_t ws_size,
                              hipStream_t stream)
{
    const float* x       = (const float*)d_in[0];
    const float* l0_side = (const float*)d_in[1];
    const float* l0_mid  = (const float*)d_in[2];
    const float* l1_side = (const float*)d_in[3];
    const float* l1_mid  = (const float*)d_in[4];
    const float* l2_side = (const float*)d_in[5];
    const float* l2_mid  = (const float*)d_in[6];
    const float* lf_side = (const float*)d_in[7];
    const float* lf_mid  = (const float*)d_in[8];
    float* out = (float*)d_out;
    char*  ws  = (char*)d_ws;

    float* q   = (float*)(ws + 0);                 // (128,28,28,2) f32
    float* chA = (float*)(ws + 1*(1 << 20));       // channel ping-pong buffers
    float* chB = (float*)(ws + 2*(1 << 20));
    float* chC = (float*)(ws + 3*(1 << 20));
    float* chD = (float*)(ws + 4*(1 << 20));

    int nImg = 128*28*28;
    tn_embed_kernel<<<(nImg + 255)/256, 256, 0, stream>>>(x, q, nImg);

    // layer 0: channels=[q] (NC=2); side stride 8*2*576, mid stride 2*576*2
    {
        int npix = 128*26*26;
        tn_string_kernel<2,2,0><<<npix/NP, BLOCK, LDS_TOTAL, stream>>>(
            q, q, l0_side,           l0_mid,        chA, 28, 28, 0, 1.f);
        tn_string_kernel<2,2,0><<<npix/NP, BLOCK, LDS_TOTAL, stream>>>(
            q, q, l0_side + 8*2*576, l0_mid + 2304, chB, 28, 28, 1, 1.f);
    }
    // layer 1: channels=[A,B] (NC=4); side stride 8*4*576, mid stride 2*2*576*2
    {
        int npix = 128*24*24;
        tn_string_kernel<4,2,0><<<npix/NP, BLOCK, LDS_TOTAL, stream>>>(
            chA, chB, l1_side,           l1_mid,        chC, 26, 26, 0, 1.f);
        tn_string_kernel<4,2,0><<<npix/NP, BLOCK, LDS_TOTAL, stream>>>(
            chA, chB, l1_side + 8*4*576, l1_mid + 4608, chD, 26, 26, 1, 1.f);
    }
    // layer 2
    {
        int npix = 128*22*22;
        tn_string_kernel<4,2,0><<<npix/NP, BLOCK, LDS_TOTAL, stream>>>(
            chC, chD, l2_side,           l2_mid,        chA, 24, 24, 0, 1.f);
        tn_string_kernel<4,2,0><<<npix/NP, BLOCK, LDS_TOTAL, stream>>>(
            chC, chD, l2_side + 8*4*576, l2_mid + 4608, chB, 24, 24, 1, 1.f);
    }
    // final: o=10, mean over 20x20 pixels via scaled atomics
    tn_zero_kernel<<<(1280 + 255)/256, 256, 0, stream>>>(out, 1280);
    {
        int npix = 128*20*20;
        tn_string_kernel<4,10,1><<<npix/NP, BLOCK, LDS_TOTAL, stream>>>(
            chA, chB, lf_side, lf_mid, out, 22, 22, 0, 1.0f/400.0f);
    }
}
